// WGAT_29068338659497
// MI455X (gfx1250) — matrix operand order repaired and checked by reading, not compiled, not run on hardware
//
#include <hip/hip_runtime.h>
#include <cstdint>
#include <cstddef>

// ---------------------------------------------------------------------------
// Types for WMMA (CDNA5 / gfx1250, wave32)
// ---------------------------------------------------------------------------
typedef __attribute__((ext_vector_type(16))) _Float16 v16h;
typedef __attribute__((ext_vector_type(8)))  float    v8f;

#define N_NODES  50000
#define N_GRAPHS 64
#define BATCH    64
#define LRELU    0.2f
#define BN_EPS   1e-5f

#define CCHUNK 64
#define KCH    (CCHUNK * 8)         // 512 K per chunk, 16 wmma k-steps
#define KSTEPS (KCH / 32)           // 16
#define WFRAG_HALVES (KSTEPS * 2 * 32 * 16)  // 16384 halves = 32KB per chunk
#define KSPLIT 4

// async global->LDS copy of 16B per lane (CDNA5 GLOBAL_LOAD_ASYNC_TO_LDS_B128)
__device__ __forceinline__ void async_b128(unsigned lds_off, unsigned long long gaddr) {
  asm volatile("global_load_async_to_lds_b128 %0, %1, off"
               :: "v"(lds_off), "v"(gaddr) : "memory");
}
template <typename T>
__device__ __forceinline__ unsigned lds_off32(T* p) {
  return (unsigned)(size_t)p;  // flat LDS addr[31:0] == group-segment offset
}

// ---------------------------------------------------------------------------
// Pre-pack kernels: f32 -> f16 embedding tables; conv weights pre-swizzled
// into per-lane B-fragment order (zero-padded per 64-cin chunk).
// ---------------------------------------------------------------------------
__global__ void cvt_f16_kernel(const float* __restrict__ src, _Float16* __restrict__ dst, int n)
{
  int t = blockIdx.x * blockDim.x + threadIdx.x;
  if (t < n) dst[t] = (_Float16)src[t];
}

__global__ void pack_w_kernel(const float* __restrict__ w, _Float16* __restrict__ dst,
                              int CIN, int total)
{
  int t = blockIdx.x * blockDim.x + threadIdx.x;
  if (t >= total) return;
  int j    = t & 15;
  int ln   = (t >> 4) & 31;
  int tile = (t >> 9) & 1;
  int ks   = (t >> 10) & 15;
  int ch   = t >> 14;
  int K    = (j & 7) + (ln >> 4) * 8 + (j >> 3) * 16;  // B 32x16: lanes0-15 K=0..7,16..23; lanes16-31 K=8..15,24..31
  int kl   = ks * 32 + K;
  int cin  = ch * CCHUNK + (kl >> 3), k = kl & 7;
  int oc   = (ln & 15) + tile * 16;
  dst[t] = (cin < CIN) ? (_Float16)w[(oc * CIN + cin) * 8 + k] : (_Float16)0.f;
}

// ---------------------------------------------------------------------------
// Conv1d branches as implicit GEMM with v_wmma_f32_16x16x32_f16.
// One block = (batch b, branch br, K-split z). M = l (121->128 pad), N = oc(32),
// K = cin*8, chunked 64 cin (512 K). Double-buffered LDS, staged with
// async global->LDS b128 copies; partial sums atomically accumulated.
// ---------------------------------------------------------------------------
__global__ __launch_bounds__(256) void conv_wmma_kernel(
    const int* __restrict__ idxA, const int* __restrict__ idxB,
    const _Float16* __restrict__ embA, const _Float16* __restrict__ embB,
    const _Float16* __restrict__ wpkA, const _Float16* __restrict__ wpkB,
    const _Float16* __restrict__ zrow,
    int cinA, int cinB,
    float* __restrict__ outA, float* __restrict__ outB)
{
  const int b  = blockIdx.x;
  const int br = blockIdx.y;
  const int z  = blockIdx.z;
  const int*      idx   = br ? idxB  : idxA;
  const _Float16* emb16 = br ? embB  : embA;
  const _Float16* wpk   = br ? wpkB  : wpkA;
  const int       CIN   = br ? cinB  : cinA;
  float*          out   = br ? outB  : outA;

  // double-buffered: x rows (64 cin x 136) + B fragments (per-lane order)
  __shared__ __align__(16) _Float16 xs[2][CCHUNK][136];
  __shared__ __align__(16) _Float16 wf[2][KSTEPS][2][32][16];

  const int tid      = threadIdx.x;
  const int lane     = tid & 31;
  const int wave     = tid >> 5;
  const int half_sel = lane >> 4;
  const int row      = lane & 15;
  const int l_base   = wave * 16;

  // zero the 8-half slide-over tails of both buffers once
  for (int i = tid; i < 2 * CCHUNK * 8; i += 256) {
    int s = i / (CCHUNK * 8), rr = (i >> 3) % CCHUNK, d = i & 7;
    xs[s][rr][128 + d] = (_Float16)0.f;
  }

  v8f acc0 = {};
  v8f acc1 = {};

  const int nchunks = (CIN + CCHUNK - 1) / CCHUNK;     // 47
  const int per     = (nchunks + KSPLIT - 1) / KSPLIT;
  const int ch0     = z * per;
  const int ch1     = (ch0 + per < nchunks) ? (ch0 + per) : nchunks;

  // issue 12 async b128 loads per lane for chunk ch into buffer sel
  auto stage = [&](int ch, int sel) {
    // weights: contiguous 32KB chunk from pre-packed buffer
    const char* wsrc = (const char*)(wpk + (size_t)ch * WFRAG_HALVES);
    unsigned    wdst = lds_off32(&wf[sel][0][0][0][0]);
    #pragma unroll
    for (int it = 0; it < 8; ++it) {
      int o = (tid + it * 256) * 16;
      async_b128(wdst + o, (unsigned long long)(wsrc + o));
    }
    // x rows: 4 lanes per row, 64B each, gathered from f16 emb (or zero page)
    int r = tid >> 2, part = tid & 3;
    int cin = ch * CCHUNK + r;
    const char* src;
    if (cin < CIN) {
      int e = idx[b * CIN + cin];
      src = (const char*)(emb16 + (size_t)e * 128);
    } else {
      src = (const char*)zrow;
    }
    unsigned xdst = lds_off32(&xs[sel][r][0]) + part * 64;
    src += part * 64;
    #pragma unroll
    for (int q = 0; q < 4; ++q)
      async_b128(xdst + q * 16, (unsigned long long)(src + q * 16));
  };

  stage(ch0, 0);
  int sel = 0;
  for (int ch = ch0; ch < ch1; ++ch) {
    const bool hasnext = (ch + 1) < ch1;
    if (hasnext) {
      stage(ch + 1, sel ^ 1);
      asm volatile("s_wait_asynccnt 12" ::: "memory");  // in-order: prev stage done
    } else {
      asm volatile("s_wait_asynccnt 0" ::: "memory");
    }
    __syncthreads();

    #pragma unroll 4
    for (int ks = 0; ks < KSTEPS; ++ks) {
      // A 16x32 f16 (ISA 7.12.2): two contiguous 8-half runs per lane
      v16h a;
      const _Float16* xr0 = &xs[sel][ks * 4 + half_sel][l_base + row];
      const _Float16* xr1 = &xs[sel][ks * 4 + half_sel + 2][l_base + row];
      #pragma unroll
      for (int j = 0; j < 8; ++j) { a[j] = xr0[j]; a[8 + j] = xr1[j]; }

      v16h b0 = *(const v16h*)&wf[sel][ks][0][lane][0];
      v16h b1 = *(const v16h*)&wf[sel][ks][1][lane][0];

      acc0 = __builtin_amdgcn_wmma_f32_16x16x32_f16(false, a, false, b0,
                                                    (short)0, acc0, false, false);
      acc1 = __builtin_amdgcn_wmma_f32_16x16x32_f16(false, a, false, b1,
                                                    (short)0, acc1, false, false);
    }
    __syncthreads();   // all waves done reading before next stage overwrites
    sel ^= 1;
  }

  // D 16x16 f32 layout: lane n=lane&15, VGPR v -> M = v + 8*half_sel
  #pragma unroll
  for (int v = 0; v < 8; ++v) {
    int m = v + half_sel * 8;
    int l = l_base + m;
    if (l < 121) {
      int oc = row;
      atomicAdd(&out[b * 3872 + oc * 121 + l],        acc0[v]);
      atomicAdd(&out[b * 3872 + (oc + 16) * 121 + l], acc1[v]);
    }
  }
}

// xc_rna[b,j] = sum_f 0.5*(c1+c2+bias1[oc]+bias2[oc])[b,f] * W[j,f] + fcb[j]
__global__ void fc_rna_kernel(const float* __restrict__ c1, const float* __restrict__ c2,
                              const float* __restrict__ cb1, const float* __restrict__ cb2,
                              const float* __restrict__ W, const float* __restrict__ bias,
                              float* __restrict__ out)
{
  int t = blockIdx.x * blockDim.x + threadIdx.x;
  if (t >= BATCH * 128) return;
  int b = t >> 7, j = t & 127;
  const float* p1 = c1 + b * 3872;
  const float* p2 = c2 + b * 3872;
  const float* w  = W + j * 3872;
  float s = 0.f;
  for (int oc = 0; oc < 32; ++oc) {
    float bb = 0.5f * (cb1[oc] + cb2[oc]);
    int base = oc * 121;
    for (int l = 0; l < 121; ++l) {
      int f = base + l;
      s += (0.5f * (p1[f] + p2[f]) + bb) * w[f];
    }
  }
  out[t] = s + bias[j];
}

// ---------------------------------------------------------------------------
// GCN
// ---------------------------------------------------------------------------
__global__ void deg_kernel(const int* __restrict__ ei, const float* __restrict__ w,
                           float* __restrict__ deg, int E, int N)
{
  int t = blockIdx.x * blockDim.x + threadIdx.x;
  if (t >= E + N) return;
  if (t < E) atomicAdd(&deg[ei[E + t]], w[t]);
  else       atomicAdd(&deg[t - E], 1.0f);
}

__global__ void dinv_kernel(float* __restrict__ deg, int N)
{
  int t = blockIdx.x * blockDim.x + threadIdx.x;
  if (t >= N) return;
  float d = deg[t];
  deg[t] = (d > 0.f) ? rsqrtf(d) : 0.f;
}

__global__ void gcn_h_kernel(const float* __restrict__ x, const float* __restrict__ W,
                             float* __restrict__ h, int N)
{
  int t = blockIdx.x * blockDim.x + threadIdx.x;
  if (t >= N * 33) return;
  int i = t / 33, c = t % 33;
  const float* xi = x + i * 33;
  const float* wc = W + c * 33;
  float s = 0.f;
  for (int c2 = 0; c2 < 33; ++c2) s += xi[c2] * wc[c2];
  h[t] = s;
}

__global__ void gcn_scatter_kernel(const int* __restrict__ ei, const float* __restrict__ w,
                                   const float* __restrict__ dinv, const float* __restrict__ h,
                                   float* __restrict__ xp, int E, int N)
{
  int t = blockIdx.x * blockDim.x + threadIdx.x;
  if (t >= E + N) return;
  int s_, d_; float wt;
  if (t < E) { s_ = ei[t]; d_ = ei[E + t]; wt = w[t]; }
  else       { s_ = d_ = t - E; wt = 1.f; }
  float norm = dinv[s_] * wt * dinv[d_];
  const float* hs = h + s_ * 33;
  float* xd = xp + d_ * 33;
  for (int c = 0; c < 33; ++c) atomicAdd(&xd[c], hs[c] * norm);
}

__global__ void gcn_post_kernel(float* __restrict__ xp, const float* __restrict__ gb,
                                const float* __restrict__ bm, const float* __restrict__ bv,
                                const float* __restrict__ bw, const float* __restrict__ bb, int N)
{
  int t = blockIdx.x * blockDim.x + threadIdx.x;
  if (t >= N * 33) return;
  int c = t % 33;
  float v = xp[t] + gb[c];
  v = v > 0.f ? v : 0.f;
  float inv = bw[c] * rsqrtf(bv[c] + BN_EPS);
  xp[t] = v * inv + (bb[c] - bm[c] * inv);
}

// ---------------------------------------------------------------------------
// GATv2 (heads=2, out=66)
// ---------------------------------------------------------------------------
__device__ inline unsigned fmap(float f) {
  unsigned u = __builtin_bit_cast(unsigned, f);
  return (u & 0x80000000u) ? ~u : (u | 0x80000000u);
}
__device__ inline float funmap(unsigned m) {
  unsigned u = (m & 0x80000000u) ? (m ^ 0x80000000u) : ~m;
  return __builtin_bit_cast(float, u);
}

__global__ void gat_lin_kernel(const float* __restrict__ xp,
                               const float* __restrict__ wl, const float* __restrict__ wr,
                               float* __restrict__ xl, float* __restrict__ xr, int N)
{
  int t = blockIdx.x * blockDim.x + threadIdx.x;
  if (t >= N * 132) return;
  int i = t / 132, j = t % 132;
  const float* xi = xp + i * 33;
  const float* a = wl + j * 33;
  const float* b = wr + j * 33;
  float sl = 0.f, sr = 0.f;
  for (int c = 0; c < 33; ++c) { float xv = xi[c]; sl += xv * a[c]; sr += xv * b[c]; }
  xl[t] = sl; xr[t] = sr;
}

__global__ void gat_edge1_kernel(const int* __restrict__ ei,
                                 const float* __restrict__ xl, const float* __restrict__ xr,
                                 const float* __restrict__ att,
                                 float* __restrict__ ebuf, unsigned* __restrict__ emaxu,
                                 int E, int N)
{
  int t = blockIdx.x * blockDim.x + threadIdx.x;
  if (t >= (E + N) * 2) return;
  int e = t >> 1, h = t & 1;
  int s_, d_;
  if (e < E) { s_ = ei[e]; d_ = ei[E + e]; } else { s_ = d_ = e - E; }
  const float* ls = xl + s_ * 132 + h * 66;
  const float* rd = xr + d_ * 132 + h * 66;
  const float* at = att + h * 66;
  float sum = 0.f;
  for (int c = 0; c < 66; ++c) {
    float v = ls[c] + rd[c];
    v = v > 0.f ? v : LRELU * v;
    sum += v * at[c];
  }
  ebuf[t] = sum;
  atomicMax(&emaxu[d_ * 2 + h], fmap(sum));
}

__global__ void gat_edge2_kernel(const int* __restrict__ ei, float* __restrict__ ebuf,
                                 const unsigned* __restrict__ emaxu,
                                 float* __restrict__ asum, int E, int N)
{
  int t = blockIdx.x * blockDim.x + threadIdx.x;
  if (t >= (E + N) * 2) return;
  int e = t >> 1, h = t & 1;
  int d_ = (e < E) ? ei[E + e] : (e - E);
  float a = __expf(ebuf[t] - funmap(emaxu[d_ * 2 + h]));
  ebuf[t] = a;
  atomicAdd(&asum[d_ * 2 + h], a);
}

__global__ void gat_edge3_kernel(const int* __restrict__ ei, const float* __restrict__ ebuf,
                                 const float* __restrict__ asum, const float* __restrict__ xl,
                                 float* __restrict__ out2, int E, int N)
{
  int t = blockIdx.x * blockDim.x + threadIdx.x;
  if (t >= (E + N) * 2) return;
  int e = t >> 1, h = t & 1;
  int s_, d_;
  if (e < E) { s_ = ei[e]; d_ = ei[E + e]; } else { s_ = d_ = e - E; }
  float alpha = ebuf[t] / asum[d_ * 2 + h];
  const float* ls = xl + s_ * 132 + h * 66;
  float* od = out2 + d_ * 132 + h * 66;
  for (int c = 0; c < 66; ++c) atomicAdd(&od[c], ls[c] * alpha);
}

__global__ void cnt_kernel(const int* __restrict__ batch, float* __restrict__ cnt, int N)
{
  int t = blockIdx.x * blockDim.x + threadIdx.x;
  if (t >= N) return;
  atomicAdd(&cnt[batch[t]], 1.f);
}

__global__ void gat_post_pool_kernel(const float* __restrict__ out2,
                                     const float* __restrict__ gb,
                                     const int* __restrict__ batch,
                                     float* __restrict__ pool, int N)
{
  int t = blockIdx.x * blockDim.x + threadIdx.x;
  if (t >= N * 132) return;
  int i = t / 132, j = t % 132;
  float v = out2[t] + gb[j];
  v = v > 0.f ? v : 0.f;
  atomicAdd(&pool[batch[i] * 132 + j], v);
}

__global__ void pool_div_kernel(float* __restrict__ pool, const float* __restrict__ cnt)
{
  int t = blockIdx.x * blockDim.x + threadIdx.x;
  if (t >= N_GRAPHS * 132) return;
  pool[t] /= fmaxf(cnt[t / 132], 1.f);
}

__global__ void mlp1_kernel(const float* __restrict__ pool, const float* __restrict__ w,
                            const float* __restrict__ b, float* __restrict__ m1)
{
  int t = blockIdx.x * blockDim.x + threadIdx.x;
  if (t >= N_GRAPHS * 1024) return;
  int g = t / 1024, j = t % 1024;
  const float* p  = pool + g * 132;
  const float* wj = w + j * 132;
  float s = 0.f;
  for (int c = 0; c < 132; ++c) s += p[c] * wj[c];
  s += b[j];
  m1[t] = s > 0.f ? s : 0.f;
}

__global__ void mlp2_kernel(const float* __restrict__ m1, const float* __restrict__ w,
                            const float* __restrict__ b, float* __restrict__ out)
{
  int t = blockIdx.x * blockDim.x + threadIdx.x;
  if (t >= N_GRAPHS * 128) return;
  int g = t >> 7, j = t & 127;
  const float* p  = m1 + g * 1024;
  const float* wj = w + j * 1024;
  float s = 0.f;
  for (int c = 0; c < 1024; ++c) s += p[c] * wj[c];
  out[t] = s + b[j];
}

// ---------------------------------------------------------------------------
// Launch
// ---------------------------------------------------------------------------
extern "C" void kernel_launch(void* const* d_in, const int* in_sizes, int n_in,
                              void* d_out, int out_size, void* d_ws, size_t ws_size,
                              hipStream_t stream)
{
  const int*   global_rna = (const int*)  d_in[0];
  const int*   local_rna  = (const int*)  d_in[1];
  const float* pro_x      = (const float*)d_in[2];
  const int*   edge_index = (const int*)  d_in[3];
  const float* pro_weight = (const float*)d_in[4];
  const int*   pro_batch  = (const int*)  d_in[5];
  const float* emb1       = (const float*)d_in[6];
  const float* emb2       = (const float*)d_in[7];
  const float* conv1_w    = (const float*)d_in[8];
  const float* conv1_b    = (const float*)d_in[9];
  const float* conv2_w    = (const float*)d_in[10];
  const float* conv2_b    = (const float*)d_in[11];
  const float* fc_xr_w    = (const float*)d_in[12];
  const float* fc_xr_b    = (const float*)d_in[13];
  const float* gcn_w      = (const float*)d_in[14];
  const float* gcn_b      = (const float*)d_in[15];
  const float* bn_mean    = (const float*)d_in[16];
  const float* bn_var     = (const float*)d_in[17];
  const float* bn_weight  = (const float*)d_in[18];
  const float* bn_bias    = (const float*)d_in[19];
  const float* gat_wl     = (const float*)d_in[20];
  const float* gat_wr     = (const float*)d_in[21];
  const float* gat_att    = (const float*)d_in[22];
  const float* gat_b      = (const float*)d_in[23];
  const float* fc1_w      = (const float*)d_in[24];
  const float* fc1_b      = (const float*)d_in[25];
  const float* fc2_w      = (const float*)d_in[26];
  const float* fc2_b      = (const float*)d_in[27];

  const int CIN1 = in_sizes[0] / BATCH;       // 3000
  const int CIN2 = in_sizes[1] / BATCH;       // 2998
  const int E    = in_sizes[3] / 2;           // 800000
  const int N    = in_sizes[5];               // 50000
  const int E2   = E + N;
  const int NCH1 = (CIN1 + CCHUNK - 1) / CCHUNK;   // 47
  const int NCH2 = (CIN2 + CCHUNK - 1) / CCHUNK;   // 47
  const int NEMB1 = in_sizes[6];              // 5*128
  const int NEMB2 = in_sizes[7];              // 65*128

  // ---- workspace carve-up ----
  char* base = (char*)d_ws;
  size_t off = 0;
  auto take = [&](size_t bytes) -> char* {
    char* p = base + off;
    off = (off + bytes + 255) & ~(size_t)255;
    return p;
  };
  float*     c1    = (float*)    take((size_t)BATCH * 3872 * 4);
  float*     c2    = (float*)    take((size_t)BATCH * 3872 * 4);
  _Float16*  emb16a= (_Float16*) take((size_t)NEMB1 * 2);
  _Float16*  emb16b= (_Float16*) take((size_t)NEMB2 * 2);
  _Float16*  wpk1  = (_Float16*) take((size_t)NCH1 * WFRAG_HALVES * 2);
  _Float16*  wpk2  = (_Float16*) take((size_t)NCH2 * WFRAG_HALVES * 2);
  _Float16*  zrow  = (_Float16*) take(256);
  float*     hbuf  = (float*)    take((size_t)N * 33 * 4);
  float*     deg   = (float*)    take((size_t)N * 4);
  float*     xp1   = (float*)    take((size_t)N * 33 * 4);
  float*     xl    = (float*)    take((size_t)N * 132 * 4);
  float*     xr    = (float*)    take((size_t)N * 132 * 4);
  float*     ebuf  = (float*)    take((size_t)E2 * 2 * 4);
  unsigned*  emaxu = (unsigned*) take((size_t)N * 2 * 4);
  float*     asum  = (float*)    take((size_t)N * 2 * 4);
  float*     out2  = (float*)    take((size_t)N * 132 * 4);
  float*     pool  = (float*)    take((size_t)N_GRAPHS * 132 * 4);
  float*     cnt   = (float*)    take((size_t)N_GRAPHS * 4);
  float*     m1    = (float*)    take((size_t)N_GRAPHS * 1024 * 4);

  float* out_rna = (float*)d_out;            // [64,128]
  float* out_pro = (float*)d_out + 8192;     // [64,128]

  // ---- zero accumulation buffers ----
  hipMemsetAsync(c1,    0, (size_t)BATCH * 3872 * 4,     stream);
  hipMemsetAsync(c2,    0, (size_t)BATCH * 3872 * 4,     stream);
  hipMemsetAsync(zrow,  0, 256,                          stream);
  hipMemsetAsync(deg,   0, (size_t)N * 4,                stream);
  hipMemsetAsync(xp1,   0, (size_t)N * 33 * 4,           stream);
  hipMemsetAsync(emaxu, 0, (size_t)N * 2 * 4,            stream);
  hipMemsetAsync(asum,  0, (size_t)N * 2 * 4,            stream);
  hipMemsetAsync(out2,  0, (size_t)N * 132 * 4,          stream);
  hipMemsetAsync(pool,  0, (size_t)N_GRAPHS * 132 * 4,   stream);
  hipMemsetAsync(cnt,   0, (size_t)N_GRAPHS * 4,         stream);

  const int T = 256;
  auto nb = [](long n, int t) { return (int)((n + t - 1) / t); };

  // ---- pre-pack f16 operands ----
  cvt_f16_kernel<<<nb(NEMB1, T), T, 0, stream>>>(emb1, emb16a, NEMB1);
  cvt_f16_kernel<<<nb(NEMB2, T), T, 0, stream>>>(emb2, emb16b, NEMB2);
  pack_w_kernel<<<nb((long)NCH1 * WFRAG_HALVES, T), T, 0, stream>>>(
      conv1_w, wpk1, CIN1, NCH1 * WFRAG_HALVES);
  pack_w_kernel<<<nb((long)NCH2 * WFRAG_HALVES, T), T, 0, stream>>>(
      conv2_w, wpk2, CIN2, NCH2 * WFRAG_HALVES);

  // ---- RNA branches: async-staged WMMA implicit-GEMM conv + fused FC ----
  dim3 cgrid(BATCH, 2, KSPLIT);
  conv_wmma_kernel<<<cgrid, 256, 0, stream>>>(
      global_rna, local_rna, emb16a, emb16b, wpk1, wpk2, zrow,
      CIN1, CIN2, c1, c2);
  fc_rna_kernel<<<nb(BATCH * 128, T), T, 0, stream>>>(c1, c2, conv1_b, conv2_b,
                                                      fc_xr_w, fc_xr_b, out_rna);

  // ---- GCN ----
  deg_kernel<<<nb(E2, T), T, 0, stream>>>(edge_index, pro_weight, deg, E, N);
  dinv_kernel<<<nb(N, T), T, 0, stream>>>(deg, N);
  gcn_h_kernel<<<nb((long)N * 33, T), T, 0, stream>>>(pro_x, gcn_w, hbuf, N);
  gcn_scatter_kernel<<<nb(E2, T), T, 0, stream>>>(edge_index, pro_weight, deg, hbuf, xp1, E, N);
  gcn_post_kernel<<<nb((long)N * 33, T), T, 0, stream>>>(xp1, gcn_b, bn_mean, bn_var,
                                                         bn_weight, bn_bias, N);

  // ---- GATv2 ----
  gat_lin_kernel<<<nb((long)N * 132, T), T, 0, stream>>>(xp1, gat_wl, gat_wr, xl, xr, N);
  gat_edge1_kernel<<<nb((long)E2 * 2, T), T, 0, stream>>>(edge_index, xl, xr, gat_att,
                                                          ebuf, emaxu, E, N);
  gat_edge2_kernel<<<nb((long)E2 * 2, T), T, 0, stream>>>(edge_index, ebuf, emaxu, asum, E, N);
  gat_edge3_kernel<<<nb((long)E2 * 2, T), T, 0, stream>>>(edge_index, ebuf, asum, xl, out2, E, N);

  // ---- pool + MLP head ----
  cnt_kernel<<<nb(N, T), T, 0, stream>>>(pro_batch, cnt, N);
  gat_post_pool_kernel<<<nb((long)N * 132, T), T, 0, stream>>>(out2, gat_b, pro_batch, pool, N);
  pool_div_kernel<<<nb(N_GRAPHS * 132, T), T, 0, stream>>>(pool, cnt);
  mlp1_kernel<<<nb(N_GRAPHS * 1024, T), T, 0, stream>>>(pool, fc1_w, fc1_b, m1);
  mlp2_kernel<<<nb(N_GRAPHS * 128, T), T, 0, stream>>>(m1, fc2_w, fc2_b, out_pro);

  (void)n_in; (void)out_size; (void)ws_size;
}
